// LocalLatentAttentionTriton_65274912965241
// MI455X (gfx1250) — matrix operand order repaired and checked by reading, not compiled, not run on hardware
//
#include <hip/hip_runtime.h>
#include <hip/hip_bf16.h>
#include <stdint.h>

// ---------------- types ----------------
typedef __bf16 bf16_t;
typedef __attribute__((ext_vector_type(16))) __bf16 v16bf;
typedef __attribute__((ext_vector_type(8)))  __bf16 v8bf;
typedef __attribute__((ext_vector_type(8)))  float  v8f;

#define D_MODEL 1024
#define N_HEADS 16
#define LATENT_HEADS 8
#define LATENT_DIM 512
#define N_LATENTS 64
#define WINDOW 256
#define HEAD_DIM 64
#define BB 2
#define SS 4096

// ---------------- scalar helpers ----------------
__device__ __forceinline__ bf16_t f2bf(float x) {
  union { float f; uint32_t u; } v; v.f = x;
  uint32_t r = v.u + 0x7FFFu + ((v.u >> 16) & 1u);   // RNE
  uint16_t h = (uint16_t)(r >> 16);
  return __builtin_bit_cast(bf16_t, h);
}
__device__ __forceinline__ float bf2f(bf16_t b) {
  uint16_t h = __builtin_bit_cast(uint16_t, b);
  union { uint32_t u; float f; } v; v.u = ((uint32_t)h) << 16;
  return v.f;
}
__device__ __forceinline__ v8f vzero8() {
  v8f v;
#pragma unroll
  for (int i = 0; i < 8; ++i) v[i] = 0.f;
  return v;
}
__device__ __forceinline__ v8f wmma_bf16(v16bf a, v16bf b, v8f c) {
  // D = A(16x32 bf16) * B(32x16 bf16) + C(16x16 f32)
  return __builtin_amdgcn_wmma_f32_16x16x32_bf16(false, a, false, b, (short)0, c,
                                                 false, false);
}
__device__ __forceinline__ v16bf cat8(v8bf lo, v8bf hi) {
  return __builtin_shufflevector(lo, hi, 0, 1, 2, 3, 4, 5, 6, 7, 8, 9, 10, 11, 12,
                                 13, 14, 15);
}

// A fragment (16x32, 16-bit): lanes 0-15 hold row=lane, K k0..k0+15;
// lanes 16-31 hold row=lane-16, K k0+16..k0+31.
__device__ __forceinline__ v16bf load_a_frag(const bf16_t* A, int lda, int k0,
                                             int lane) {
  int ln = lane & 15, hs = lane >> 4;
  const bf16_t* p = A + (size_t)ln * lda + k0 + (hs << 4);
  v8bf lo = *(const v8bf*)(p);
  v8bf hi = *(const v8bf*)(p + 8);
  return cat8(lo, hi);
}

// B fragment (32x16, 16-bit) from a PRE-TRANSPOSED weight B^T [N][K]:
// lane holds col = col0+(lane&15) (a row of B^T); lanes 0-15: K=k0..k0+15,
// lanes 16-31: K=k0+16..k0+31 -> one contiguous 32-byte load.
__device__ __forceinline__ v16bf load_b_frag_tr(const bf16_t* BT, int ldk, int k0,
                                                int col0, int lane) {
  int ln = lane & 15, hs = lane >> 4;
  const bf16_t* p = BT + (size_t)(col0 + ln) * ldk + k0 + (hs << 4);
  return *(const v16bf*)(p);
}

// ---------------- fp32 -> bf16 (straight copy) ----------------
__global__ __launch_bounds__(256) void f32_to_bf16_kernel(const float* __restrict__ s,
                                                          bf16_t* __restrict__ d,
                                                          int n) {
  int i = blockIdx.x * blockDim.x + threadIdx.x;
  if (i < n) d[i] = f2bf(s[i]);
}

// ---------------- fp32 [K,N] -> bf16 transposed [N,K] ----------------
__global__ __launch_bounds__(256) void f32_to_bf16_tr_kernel(
    const float* __restrict__ s, bf16_t* __restrict__ d, int K, int N) {
  int i = blockIdx.x * blockDim.x + threadIdx.x;
  if (i >= K * N) return;
  int k = i % K, n = i / K;                 // i indexes dst [N][K]
  d[(size_t)n * K + k] = f2bf(s[(size_t)k * N + n]);
}

// ---------------- generic WMMA GEMM: C[M,N] = A[M,K] @ B[K,N] ----------------
// B supplied pre-transposed as BT [N][K]; one wave -> 16x64 output tile.
template <int OUTF32>
__global__ __launch_bounds__(128) void gemm_bf16_kernel(const bf16_t* __restrict__ A,
                                                        const bf16_t* __restrict__ BT,
                                                        void* __restrict__ Cp, int M,
                                                        int N, int K) {
  int lane = threadIdx.x & 31;
  int wid = blockIdx.x * (blockDim.x >> 5) + (threadIdx.x >> 5);
  int ntg = N >> 6;
  int total = (M >> 4) * ntg;
  if (wid >= total) return;
  int mt = wid / ntg, ng = wid - mt * ntg;
  int row0 = mt << 4, col0 = ng << 6;
  v8f acc[4];
#pragma unroll
  for (int t = 0; t < 4; ++t) acc[t] = vzero8();
  const bf16_t* Arow = A + (size_t)row0 * K;
  for (int k0 = 0; k0 < K; k0 += 32) {
    v16bf a = load_a_frag(Arow, K, k0, lane);
#pragma unroll
    for (int t = 0; t < 4; ++t) {
      v16bf b = load_b_frag_tr(BT, K, k0, col0 + t * 16, lane);
      acc[t] = wmma_bf16(a, b, acc[t]);
    }
  }
  int ln = lane & 15, hs = lane >> 4;
  int rbase = row0 + (hs << 3);
#pragma unroll
  for (int t = 0; t < 4; ++t) {
    int c = col0 + t * 16 + ln;
#pragma unroll
    for (int r = 0; r < 8; ++r) {
      size_t off = (size_t)(rbase + r) * N + c;
      if (OUTF32)
        ((float*)Cp)[off] = acc[t][r];
      else
        ((bf16_t*)Cp)[off] = f2bf(acc[t][r]);
    }
  }
}

// ---------------- chunk-mean pooling: [B*S,512] -> [B*64,512] ----------------
__global__ __launch_bounds__(256) void pool_kernel(const bf16_t* __restrict__ src,
                                                   bf16_t* __restrict__ dst) {
  int idx = blockIdx.x * blockDim.x + threadIdx.x;
  const int total = BB * N_LATENTS * LATENT_DIM;
  if (idx >= total) return;
  int c = idx % LATENT_DIM;
  int lat = (idx / LATENT_DIM) % N_LATENTS;
  int b = idx / (LATENT_DIM * N_LATENTS);
  const bf16_t* p = src + ((size_t)b * SS + (size_t)lat * 64) * LATENT_DIM + c;
  float s = 0.f;
#pragma unroll 8
  for (int j = 0; j < 64; ++j) s += bf2f(p[(size_t)j * LATENT_DIM]);
  dst[idx] = f2bf(s * (1.0f / 64.0f));
}

// ---------------- local sliding-window flash attention ----------------
// one wave per (b, h, 16-query tile); 32-key steps; online softmax.
__global__ __launch_bounds__(128) void local_attn_kernel(
    const bf16_t* __restrict__ Q, const bf16_t* __restrict__ Km,
    const bf16_t* __restrict__ V, bf16_t* __restrict__ localo) {
  __shared__ bf16_t plds[4][16 * 32];
  int lane = threadIdx.x & 31;
  int w = threadIdx.x >> 5;
  int wid = blockIdx.x * 4 + w;
  const int QT = SS / 16;
  if (wid >= BB * N_HEADS * QT) return;
  int qt = wid % QT;
  int h = (wid / QT) % N_HEADS;
  int b = wid / (QT * N_HEADS);
  int q0 = qt * 16;
  int ln = lane & 15, hs = lane >> 4, rbase = hs * 8;

  // Q A-fragments (16 rows x 64 dims -> two K=32 steps)
  size_t qbase = ((size_t)b * SS + q0 + ln) * (size_t)D_MODEL + h * HEAD_DIM;
  v16bf aq[2];
#pragma unroll
  for (int t = 0; t < 2; ++t) {
    int kb = t * 32 + (hs << 4);
    v8bf lo = *(const v8bf*)(Q + qbase + kb);
    v8bf hi = *(const v8bf*)(Q + qbase + kb + 8);
    aq[t] = cat8(lo, hi);
  }

  float m[8], l[8];
  v8f o[4];
#pragma unroll
  for (int r = 0; r < 8; ++r) { m[r] = -1e30f; l[r] = 0.f; }
#pragma unroll
  for (int t = 0; t < 4; ++t) o[t] = vzero8();

  int ks = q0 - (WINDOW - 1);
  if (ks < 0) ks = 0;
  ks &= ~31;
  int kend = q0 + 16;
  const float scale = 0.125f;  // 1/sqrt(64)
  bf16_t* pl = &plds[w][0];

  for (int kt = ks; kt < kend; kt += 32) {
    // ---- scores: two 16x16 tiles (keys kt..kt+15 and kt+16..kt+31) ----
    v8f s0 = vzero8(), s1 = vzero8();
    int key0 = kt + ln;        if (key0 > SS - 1) key0 = SS - 1;
    int key1 = kt + 16 + ln;   if (key1 > SS - 1) key1 = SS - 1;
    const bf16_t* kp0 = Km + ((size_t)b * SS + key0) * D_MODEL + h * HEAD_DIM;
    const bf16_t* kp1 = Km + ((size_t)b * SS + key1) * D_MODEL + h * HEAD_DIM;
#pragma unroll
    for (int t = 0; t < 2; ++t) {
      int kb = t * 32 + (hs << 4);
      v16bf bk0 = *(const v16bf*)(kp0 + kb);
      v16bf bk1 = *(const v16bf*)(kp1 + kb);
      s0 = wmma_bf16(aq[t], bk0, s0);
      s1 = wmma_bf16(aq[t], bk1, s1);
    }
    // ---- mask + online softmax (rows live across 16-lane groups) ----
    float p0[8], p1[8];
#pragma unroll
    for (int r = 0; r < 8; ++r) {
      int q = q0 + rbase + r;
      int k0i = kt + ln, k1i = kt + 16 + ln;
      float v0 = ((k0i <= q) && (k0i >= q - (WINDOW - 1))) ? s0[r] * scale : -1e30f;
      float v1 = ((k1i <= q) && (k1i >= q - (WINDOW - 1))) ? s1[r] * scale : -1e30f;
      float t = fmaxf(v0, v1);
      t = fmaxf(t, __shfl_xor(t, 1, 32));
      t = fmaxf(t, __shfl_xor(t, 2, 32));
      t = fmaxf(t, __shfl_xor(t, 4, 32));
      t = fmaxf(t, __shfl_xor(t, 8, 32));
      float mn = fmaxf(m[r], t);
      float so = __expf(m[r] - mn);
      p0[r] = (v0 <= -1e29f) ? 0.f : __expf(v0 - mn);
      p1[r] = (v1 <= -1e29f) ? 0.f : __expf(v1 - mn);
      float rs = p0[r] + p1[r];
      rs += __shfl_xor(rs, 1, 32);
      rs += __shfl_xor(rs, 2, 32);
      rs += __shfl_xor(rs, 4, 32);
      rs += __shfl_xor(rs, 8, 32);
      l[r] = l[r] * so + rs;
      m[r] = mn;
#pragma unroll
      for (int t2 = 0; t2 < 4; ++t2) o[t2][r] = o[t2][r] * so;
    }
    // ---- stage P (C-layout -> row-major LDS -> A-layout) ----
#pragma unroll
    for (int r = 0; r < 8; ++r) {
      pl[(rbase + r) * 32 + ln] = f2bf(p0[r]);
      pl[(rbase + r) * 32 + 16 + ln] = f2bf(p1[r]);
    }
    asm volatile("s_wait_dscnt 0" ::: "memory");
    v16bf pa;
    {
      const bf16_t* pr = pl + ln * 32 + (hs << 4);
      v8bf lo = *(const v8bf*)(pr);
      v8bf hi = *(const v8bf*)(pr + 8);
      pa = cat8(lo, hi);
    }
    // ---- O += P @ V (keys kt..kt+31) ----
#pragma unroll
    for (int t = 0; t < 4; ++t) {
      int col = h * HEAD_DIM + t * 16 + ln;
      int kbase = kt + (hs << 4);
      v16bf bv;
#pragma unroll
      for (int i = 0; i < 16; ++i) {
        int key = kbase + i;
        if (key > SS - 1) key = SS - 1;  // p==0 there
        bv[i] = V[((size_t)b * SS + key) * D_MODEL + col];
      }
      o[t] = wmma_bf16(pa, bv, o[t]);
    }
    asm volatile("s_wait_dscnt 0" ::: "memory");  // LDS reuse next iter
  }
  // ---- normalize + store local[b][h][s][d] ----
#pragma unroll
  for (int t = 0; t < 4; ++t) {
#pragma unroll
    for (int r = 0; r < 8; ++r) {
      float v = o[t][r] / l[r];
      size_t off = (((size_t)(b * N_HEADS + h)) * SS + q0 + rbase + r) * HEAD_DIM +
                   t * 16 + ln;
      localo[off] = f2bf(v);
    }
  }
}

// ---------------- latent attention (64 latents, chunk-causal) ----------------
__global__ __launch_bounds__(128) void latent_attn_kernel(
    const bf16_t* __restrict__ GQ, const bf16_t* __restrict__ GKp,
    const bf16_t* __restrict__ GVp, bf16_t* __restrict__ rem) {
  __shared__ bf16_t plds[4][16 * 64];
  int lane = threadIdx.x & 31;
  int w = threadIdx.x >> 5;
  int wid = blockIdx.x * 4 + w;
  const int QT = SS / 16;
  if (wid >= BB * LATENT_HEADS * QT) return;
  int qt = wid % QT;
  int lh = (wid / QT) % LATENT_HEADS;
  int b = wid / (QT * LATENT_HEADS);
  int q0 = qt * 16;
  int ln = lane & 15, hs = lane >> 4, rbase = hs * 8;

  size_t qbase = ((size_t)b * SS + q0 + ln) * (size_t)LATENT_DIM + lh * 64;
  v16bf aq[2];
#pragma unroll
  for (int t = 0; t < 2; ++t) {
    int kb = t * 32 + (hs << 4);
    v8bf lo = *(const v8bf*)(GQ + qbase + kb);
    v8bf hi = *(const v8bf*)(GQ + qbase + kb + 8);
    aq[t] = cat8(lo, hi);
  }
  // scores vs all 64 latents
  v8f sc[4];
#pragma unroll
  for (int ct = 0; ct < 4; ++ct) {
    sc[ct] = vzero8();
    int lat = ct * 16 + ln;
    const bf16_t* kp = GKp + ((size_t)b * N_LATENTS + lat) * LATENT_DIM + lh * 64;
#pragma unroll
    for (int t = 0; t < 2; ++t) {
      int kb = t * 32 + (hs << 4);
      v16bf bk = *(const v16bf*)(kp + kb);
      sc[ct] = wmma_bf16(aq[t], bk, sc[ct]);
    }
  }
  int nvalid = q0 / 64 + 1;  // tile-uniform chunk-causal mask
  const float scale = 0.125f;
  float p[4][8];
#pragma unroll
  for (int r = 0; r < 8; ++r) {
    float vv[4];
    float mv = -1e30f;
#pragma unroll
    for (int ct = 0; ct < 4; ++ct) {
      int lat = ct * 16 + ln;
      vv[ct] = (lat < nvalid) ? sc[ct][r] * scale : -1e30f;
      mv = fmaxf(mv, vv[ct]);
    }
    mv = fmaxf(mv, __shfl_xor(mv, 1, 32));
    mv = fmaxf(mv, __shfl_xor(mv, 2, 32));
    mv = fmaxf(mv, __shfl_xor(mv, 4, 32));
    mv = fmaxf(mv, __shfl_xor(mv, 8, 32));
    float sum = 0.f;
#pragma unroll
    for (int ct = 0; ct < 4; ++ct) {
      p[ct][r] = (vv[ct] <= -1e29f) ? 0.f : __expf(vv[ct] - mv);
      sum += p[ct][r];
    }
    sum += __shfl_xor(sum, 1, 32);
    sum += __shfl_xor(sum, 2, 32);
    sum += __shfl_xor(sum, 4, 32);
    sum += __shfl_xor(sum, 8, 32);
    float inv = 1.f / sum;
#pragma unroll
    for (int ct = 0; ct < 4; ++ct) p[ct][r] *= inv;
  }
  // stage P (16x64) -> LDS -> A-frags
  bf16_t* pl = &plds[w][0];
#pragma unroll
  for (int r = 0; r < 8; ++r)
#pragma unroll
    for (int ct = 0; ct < 4; ++ct)
      pl[(rbase + r) * 64 + ct * 16 + ln] = f2bf(p[ct][r]);
  asm volatile("s_wait_dscnt 0" ::: "memory");
  v16bf pa[2];
#pragma unroll
  for (int t = 0; t < 2; ++t) {
    const bf16_t* pr = pl + ln * 64 + t * 32 + (hs << 4);
    v8bf lo = *(const v8bf*)(pr);
    v8bf hi = *(const v8bf*)(pr + 8);
    pa[t] = cat8(lo, hi);
  }
  // rem = P @ GV
  v8f o[4];
#pragma unroll
  for (int ct = 0; ct < 4; ++ct) {
    o[ct] = vzero8();
    int col = lh * 64 + ct * 16 + ln;
#pragma unroll
    for (int t = 0; t < 2; ++t) {
      int kbase = t * 32 + (hs << 4);
      v16bf bv;
#pragma unroll
      for (int i = 0; i < 16; ++i)
        bv[i] = GVp[((size_t)b * N_LATENTS + kbase + i) * LATENT_DIM + col];
      o[ct] = wmma_bf16(pa[t], bv, o[ct]);
    }
  }
#pragma unroll
  for (int ct = 0; ct < 4; ++ct)
#pragma unroll
    for (int r = 0; r < 8; ++r) {
      size_t off =
          ((size_t)b * SS + q0 + rbase + r) * LATENT_DIM + lh * 64 + ct * 16 + ln;
      rem[off] = f2bf(o[ct][r]);
    }
}

// ---------------- gated combine ----------------
__global__ __launch_bounds__(256) void combine_kernel(
    const bf16_t* __restrict__ Qb, const bf16_t* __restrict__ localb,
    const bf16_t* __restrict__ rem2, const float* __restrict__ glw,
    const float* __restrict__ grw, bf16_t* __restrict__ outcat) {
  int idx = blockIdx.x * blockDim.x + threadIdx.x;
  const int total = BB * SS * N_HEADS;
  if (idx >= total) return;
  int h = idx % N_HEADS;
  int s = (idx / N_HEADS) % SS;
  int b = idx / (N_HEADS * SS);
  size_t qoff = ((size_t)b * SS + s) * D_MODEL + h * HEAD_DIM;
  float dl = 0.f, dr = 0.f;
#pragma unroll 8
  for (int d = 0; d < HEAD_DIM; ++d) {
    float q = bf2f(Qb[qoff + d]);
    dl += q * glw[h * HEAD_DIM + d];
    dr += q * grw[h * HEAD_DIM + d];
  }
  float gl = 1.f / (1.f + __expf(-dl));
  float gr = 1.f / (1.f + __expf(-dr));
  size_t loff = (((size_t)(b * N_HEADS + h)) * SS + s) * HEAD_DIM;
#pragma unroll 8
  for (int d = 0; d < HEAD_DIM; ++d) {
    float v = gl * bf2f(localb[loff + d]) + gr * bf2f(rem2[qoff + d]);
    outcat[qoff + d] = f2bf(v);
  }
}

// ---------------- host orchestration ----------------
extern "C" void kernel_launch(void* const* d_in, const int* in_sizes, int n_in,
                              void* d_out, int out_size, void* d_ws, size_t ws_size,
                              hipStream_t stream) {
  (void)in_sizes; (void)n_in; (void)out_size; (void)ws_size;
  const float* x = (const float*)d_in[0];
  const float* wq = (const float*)d_in[1];
  const float* wk = (const float*)d_in[2];
  const float* wv = (const float*)d_in[3];
  const float* wgq = (const float*)d_in[4];
  const float* wgk = (const float*)d_in[5];
  const float* wgv = (const float*)d_in[6];
  const float* r_out_w = (const float*)d_in[7];
  const float* glw = (const float*)d_in[8];
  const float* grw = (const float*)d_in[9];
  const float* w_out = (const float*)d_in[10];

  char* ws = (char*)d_ws;
  size_t off = 0;
  auto alloc_bf = [&](size_t elems) -> bf16_t* {
    bf16_t* p = (bf16_t*)(ws + off);
    off += ((elems * sizeof(bf16_t) + 255) & ~(size_t)255);
    return p;
  };
  const size_t BS = (size_t)BB * SS;
  bf16_t* xb = alloc_bf(BS * D_MODEL);
  // weights stored TRANSPOSED [N][K] for contiguous WMMA B-fragment loads
  bf16_t* wqbT = alloc_bf((size_t)D_MODEL * D_MODEL);
  bf16_t* wkbT = alloc_bf((size_t)D_MODEL * D_MODEL);
  bf16_t* wvbT = alloc_bf((size_t)D_MODEL * D_MODEL);
  bf16_t* wgqbT = alloc_bf((size_t)D_MODEL * LATENT_DIM);
  bf16_t* wgkbT = alloc_bf((size_t)D_MODEL * LATENT_DIM);
  bf16_t* wgvbT = alloc_bf((size_t)D_MODEL * LATENT_DIM);
  bf16_t* rowbT = alloc_bf((size_t)LATENT_DIM * D_MODEL);
  bf16_t* woutbT = alloc_bf((size_t)D_MODEL * D_MODEL);
  bf16_t* Qb = alloc_bf(BS * D_MODEL);
  bf16_t* Kb = alloc_bf(BS * D_MODEL);
  bf16_t* Vb = alloc_bf(BS * D_MODEL);
  bf16_t* GQb = alloc_bf(BS * LATENT_DIM);
  bf16_t* Kx = alloc_bf(BS * LATENT_DIM);
  bf16_t* Vx = alloc_bf(BS * LATENT_DIM);
  bf16_t* GKp = alloc_bf((size_t)BB * N_LATENTS * LATENT_DIM);
  bf16_t* GVp = alloc_bf((size_t)BB * N_LATENTS * LATENT_DIM);
  bf16_t* localb = alloc_bf(BS * D_MODEL);
  bf16_t* remb = alloc_bf(BS * LATENT_DIM);
  bf16_t* rem2b = alloc_bf(BS * D_MODEL);
  bf16_t* outcat = alloc_bf(BS * D_MODEL);

  auto conv = [&](const float* s, bf16_t* d, size_t n) {
    f32_to_bf16_kernel<<<dim3((unsigned)((n + 255) / 256)), dim3(256), 0, stream>>>(
        s, d, (int)n);
  };
  auto convT = [&](const float* s, bf16_t* d, int K, int N) {
    size_t n = (size_t)K * N;
    f32_to_bf16_tr_kernel<<<dim3((unsigned)((n + 255) / 256)), dim3(256), 0,
                            stream>>>(s, d, K, N);
  };
  conv(x, xb, BS * D_MODEL);
  convT(wq, wqbT, D_MODEL, D_MODEL);
  convT(wk, wkbT, D_MODEL, D_MODEL);
  convT(wv, wvbT, D_MODEL, D_MODEL);
  convT(wgq, wgqbT, D_MODEL, LATENT_DIM);
  convT(wgk, wgkbT, D_MODEL, LATENT_DIM);
  convT(wgv, wgvbT, D_MODEL, LATENT_DIM);
  convT(r_out_w, rowbT, LATENT_DIM, D_MODEL);
  convT(w_out, woutbT, D_MODEL, D_MODEL);

  auto gemm_bf = [&](const bf16_t* A, const bf16_t* BT, bf16_t* C, int M, int N,
                     int K) {
    int waves = (M / 16) * (N / 64);
    gemm_bf16_kernel<0><<<dim3((unsigned)((waves + 3) / 4)), dim3(128), 0, stream>>>(
        A, BT, (void*)C, M, N, K);
  };
  gemm_bf(xb, wqbT, Qb, (int)BS, D_MODEL, D_MODEL);
  gemm_bf(xb, wkbT, Kb, (int)BS, D_MODEL, D_MODEL);
  gemm_bf(xb, wvbT, Vb, (int)BS, D_MODEL, D_MODEL);
  gemm_bf(xb, wgqbT, GQb, (int)BS, LATENT_DIM, D_MODEL);
  gemm_bf(xb, wgkbT, Kx, (int)BS, LATENT_DIM, D_MODEL);
  gemm_bf(xb, wgvbT, Vx, (int)BS, LATENT_DIM, D_MODEL);

  {
    unsigned n = BB * N_LATENTS * LATENT_DIM;
    pool_kernel<<<dim3((n + 255) / 256), dim3(256), 0, stream>>>(Kx, GKp);
    pool_kernel<<<dim3((n + 255) / 256), dim3(256), 0, stream>>>(Vx, GVp);
  }
  {
    int waves = BB * N_HEADS * (SS / 16);
    local_attn_kernel<<<dim3((unsigned)((waves + 3) / 4)), dim3(128), 0, stream>>>(
        Qb, Kb, Vb, localb);
  }
  {
    int waves = BB * LATENT_HEADS * (SS / 16);
    latent_attn_kernel<<<dim3((unsigned)((waves + 3) / 4)), dim3(128), 0, stream>>>(
        GQb, GKp, GVp, remb);
  }
  gemm_bf(remb, rowbT, rem2b, (int)BS, D_MODEL, LATENT_DIM);
  {
    unsigned n = BB * SS * N_HEADS;
    combine_kernel<<<dim3((n + 255) / 256), dim3(256), 0, stream>>>(
        Qb, localb, rem2b, glw, grw, outcat);
  }
  {
    int waves = ((int)BS / 16) * (D_MODEL / 64);
    gemm_bf16_kernel<1><<<dim3((unsigned)((waves + 3) / 4)), dim3(128), 0, stream>>>(
        outcat, woutbT, d_out, (int)BS, D_MODEL, D_MODEL);
  }
}
